// GraphQueryProjectionGAT_10393820856963
// MI455X (gfx1250) — matrix operand order repaired and checked by reading, not compiled, not run on hardware
//
#include <hip/hip_runtime.h>
#include <hip/hip_bf16.h>

// ---------------------------------------------------------------------------
// Types for CDNA5 WMMA (gfx1250, wave32)
// ---------------------------------------------------------------------------
typedef __attribute__((ext_vector_type(16))) __bf16 v16bf;
typedef __attribute__((ext_vector_type(8)))  __bf16 v8bf;
typedef __attribute__((ext_vector_type(8)))  float  v8f;

#define LDS_PAD 40  // bf16 elems per LDS row for a 32-wide K tile (80B -> 16B-aligned frags)

// Build a 16x(2x8) bf16 fragment from two 16B-aligned LDS chunks (2x ds_load_b128).
__device__ __forceinline__ v16bf frag_from_lds(const __bf16* p0, const __bf16* p1) {
  v8bf a = *(const v8bf*)p0;
  v8bf b = *(const v8bf*)p1;
  v16bf r;
#pragma unroll
  for (int i = 0; i < 8; ++i) { r[i] = a[i]; r[8 + i] = b[i]; }
  return r;
}

// Stage 16 contiguous elements (fp32 or bf16 source) into LDS as bf16.
__device__ __forceinline__ void stage16(__bf16* dst, const float* g) {
  const float4* f4 = (const float4*)g;
  float4 a = f4[0], b = f4[1], c = f4[2], d = f4[3];
  dst[0]  = (__bf16)a.x; dst[1]  = (__bf16)a.y; dst[2]  = (__bf16)a.z; dst[3]  = (__bf16)a.w;
  dst[4]  = (__bf16)b.x; dst[5]  = (__bf16)b.y; dst[6]  = (__bf16)b.z; dst[7]  = (__bf16)b.w;
  dst[8]  = (__bf16)c.x; dst[9]  = (__bf16)c.y; dst[10] = (__bf16)c.z; dst[11] = (__bf16)c.w;
  dst[12] = (__bf16)d.x; dst[13] = (__bf16)d.y; dst[14] = (__bf16)d.z; dst[15] = (__bf16)d.w;
}
__device__ __forceinline__ void stage16(__bf16* dst, const __bf16* g) {
  uint4 u0 = ((const uint4*)g)[0];
  uint4 u1 = ((const uint4*)g)[1];
  *(uint4*)dst       = u0;
  *(uint4*)(dst + 8) = u1;
}

// ---------------------------------------------------------------------------
// K0: pack int32 adjacency -> 1 bit per edge (32 MB -> 1 MB of traffic later)
// ---------------------------------------------------------------------------
__global__ __launch_bounds__(256)
void pack_adj(const int* __restrict__ adj, unsigned* __restrict__ outb, int total) {
  int idx = blockIdx.x * 256 + threadIdx.x;
  if (idx >= total) return;
  const int* src = adj + (long)idx * 32;
  unsigned m = 0u;
#pragma unroll
  for (int j = 0; j < 32; ++j) m |= (src[j] != 0) ? (1u << j) : 0u;
  outb[idx] = m;
}

// ---------------------------------------------------------------------------
// K1: batched GEMM  C[batch] = A[batch/aDiv] (MxK) @ B[batch%bMod] (KxN=128)
// bf16 WMMA, fp32 accumulate, bf16 output. Block = 256 thr = 8 waves,
// tile 128x128x32; wave tile 32x64 (2x4 WMMA 16x16x32).
// ---------------------------------------------------------------------------
template <typename AT>
__global__ __launch_bounds__(256)
void gemm_wmma_bf16(const AT* __restrict__ A, const float* __restrict__ B,
                    __bf16* __restrict__ C, int Kdim, int aDiv, int bMod,
                    long aStride, long bStride, long cStride) {
  __shared__ __bf16 At[128 * LDS_PAD];
  __shared__ __bf16 Bt[128 * LDS_PAD];  // transposed: [n][k]

  const int batch = blockIdx.y;
  const int blockM = blockIdx.x;
  const AT*    Ab = A + (long)(batch / aDiv) * aStride;
  const float* Bb = B + (long)(batch % bMod) * bStride;
  __bf16*      Cb = C + (long)batch * cStride;

  const int t = threadIdx.x;
  const int wid = t >> 5, lane = t & 31;
  const int waveM = wid >> 1, waveN = wid & 1;   // 4 x 2 wave grid
  const int r = lane & 15, half = lane >> 4;

  v8f acc[2][4];
#pragma unroll
  for (int mi = 0; mi < 2; ++mi)
#pragma unroll
    for (int nt = 0; nt < 4; ++nt)
#pragma unroll
      for (int e = 0; e < 8; ++e) acc[mi][nt][e] = 0.f;

  const int arow = t >> 1, akseg = (t & 1) * 16;  // A: 128 rows x 32 k
  const int brow = t >> 3, bnseg = (t & 7) * 16;  // B: 32 k x 128 n

  for (int k0 = 0; k0 < Kdim; k0 += 32) {
    stage16(&At[arow * LDS_PAD + akseg],
            Ab + (long)(blockM * 128 + arow) * Kdim + k0 + akseg);
    {
      const float4* f4 = (const float4*)(Bb + (long)(k0 + brow) * 128 + bnseg);
      float4 a = f4[0], b = f4[1], c = f4[2], d = f4[3];
      float fv[16] = {a.x, a.y, a.z, a.w, b.x, b.y, b.z, b.w,
                      c.x, c.y, c.z, c.w, d.x, d.y, d.z, d.w};
#pragma unroll
      for (int i = 0; i < 16; ++i) Bt[(bnseg + i) * LDS_PAD + brow] = (__bf16)fv[i];
    }
    __syncthreads();

    v16bf bfr[4];
#pragma unroll
    for (int nt = 0; nt < 4; ++nt) {  // B 32x16: col = lane&15, kq = half*8 (+16)
      const __bf16* bp = &Bt[(waveN * 64 + nt * 16 + r) * LDS_PAD + half * 8];
      bfr[nt] = frag_from_lds(bp, bp + 16);
    }
#pragma unroll
    for (int mi = 0; mi < 2; ++mi) {  // A 16x32: row = lane&15, kq = half*8 (+16)
      const __bf16* ap = &At[(waveM * 32 + mi * 16 + r) * LDS_PAD + half * 8];
      v16bf af = frag_from_lds(ap, ap + 16);
#pragma unroll
      for (int nt = 0; nt < 4; ++nt)
        acc[mi][nt] = __builtin_amdgcn_wmma_f32_16x16x32_bf16(
            false, af, false, bfr[nt], (short)0, acc[mi][nt], false, false);
    }
    __syncthreads();
  }

#pragma unroll
  for (int mi = 0; mi < 2; ++mi)
#pragma unroll
    for (int nt = 0; nt < 4; ++nt) {
      const int colG = waveN * 64 + nt * 16 + r;
      const int rowG = blockM * 128 + waveM * 32 + mi * 16 + half * 8;
#pragma unroll
      for (int e = 0; e < 8; ++e)
        Cb[(long)(rowG + e) * 128 + colG] = (__bf16)acc[mi][nt][e];
    }
}

// ---------------------------------------------------------------------------
// K2: per-row attention scores  s1[bh][n] = h . a1[head],  s2 likewise
// ---------------------------------------------------------------------------
__global__ __launch_bounds__(256)
void gat_scores(const __bf16* __restrict__ h, const float* __restrict__ a1,
                const float* __restrict__ a2, float* __restrict__ s1,
                float* __restrict__ s2, int H, int rows) {
  int idx = blockIdx.x * 256 + threadIdx.x;
  if (idx >= rows) return;
  int head = (idx >> 10) % H;
  const __bf16* hp = h + (long)idx * 128;
  const float* a1p = a1 + head * 128;
  const float* a2p = a2 + head * 128;
  float acc1 = 0.f, acc2 = 0.f;
#pragma unroll 8
  for (int o = 0; o < 128; ++o) {
    float v = (float)hp[o];
    acc1 = fmaf(v, a1p[o], acc1);
    acc2 = fmaf(v, a2p[o], acc2);
  }
  s1[idx] = acc1;
  s2[idx] = acc2;
}

// ---------------------------------------------------------------------------
// K3: fused masked-softmax + aggregation + ELU (+ head-concat reshape)
// out[b][i][head*128+o] = ELU( softmax_j(leakyrelu(s1_i+s2_j) | adj) @ h[:,o] )
// One wave per 16-row tile; flash-style online max/sum; P@V via bf16 WMMA.
// ---------------------------------------------------------------------------
template <typename OutT>
__global__ __launch_bounds__(256)
void gat_attn(const __bf16* __restrict__ hmat, const float* __restrict__ s1v,
              const float* __restrict__ s2v, const unsigned* __restrict__ adjb,
              OutT* __restrict__ outp, int H, long outBatchStride, int outRowStride) {
  __shared__ __bf16 hT[128 * LDS_PAD];  // V tile transposed: [o][j_local]

  const int bh = blockIdx.y;
  const int b = bh / H, head = bh % H;
  const int wid = threadIdx.x >> 5, lane = threadIdx.x & 31;
  const int r = lane & 15, half = lane >> 4;
  const int i0 = blockIdx.x * 128 + wid * 16;
  const int i = i0 + r;

  const __bf16* hbh = hmat + (long)bh * 1024 * 128;
  const float* s1p = s1v + (long)bh * 1024;
  const float* s2p = s2v + (long)bh * 1024;
  const unsigned* abRow = adjb + ((long)b * 1024 + i) * 32;
  const float s1i = s1p[i];

  // ---- Phase A: online max / sum over this lane's half of the 1024 columns
  float m = -3.4e38f, l = 0.f;
  for (int w = half * 16; w < half * 16 + 16; ++w) {
    unsigned bits = abRow[w];
    if (!bits) continue;
    const int jb = w << 5;
#pragma unroll 4
    for (int jj = 0; jj < 32; ++jj) {
      if (bits & (1u << jj)) {
        float e = s1i + s2p[jb + jj];
        e = (e > 0.f) ? e : 0.2f * e;  // leaky_relu
        if (e > m) { l = l * __expf(m - e) + 1.f; m = e; }
        else       { l += __expf(e - m); }
      }
    }
  }
  {  // combine the two half-lanes of each row
    float mo = __shfl_xor(m, 16, 32);
    float lo = __shfl_xor(l, 16, 32);
    float M = fmaxf(m, mo);
    l = l * __expf(m - M) + lo * __expf(mo - M);
    m = M;
  }
  const float Mrow = m;
  const float invL = (l > 0.f) ? (1.f / l) : 0.f;

  // ---- Phase B: accumulate P (unnormalized exp, bf16) @ V with WMMA
  v8f acc[8];
#pragma unroll
  for (int nt = 0; nt < 8; ++nt)
#pragma unroll
    for (int e = 0; e < 8; ++e) acc[nt][e] = 0.f;

  const int ljj = threadIdx.x >> 3;          // cooperative V-tile staging
  const int loseg = (threadIdx.x & 7) * 16;

  for (int j0 = 0; j0 < 1024; j0 += 32) {
    {  // stage h[j0..j0+31][0..127] transposed into LDS
      const __bf16* src = hbh + (long)(j0 + ljj) * 128 + loseg;
      uint4 u0 = ((const uint4*)src)[0];
      uint4 u1 = ((const uint4*)src)[1];
      __bf16 tmp[16];
      *(uint4*)&tmp[0] = u0;
      *(uint4*)&tmp[8] = u1;
#pragma unroll
      for (int kk = 0; kk < 16; ++kk) hT[(loseg + kk) * LDS_PAD + ljj] = tmp[kk];
    }
    __syncthreads();

    // Build P A-fragment (16 rows x 32 j), ISA layout: row = lane&15, kq = half*8
    const unsigned bits = abRow[j0 >> 5];
    const int kq = half * 8;
    v16bf pf;
#pragma unroll
    for (int tt = 0; tt < 8; ++tt) {
      int jl = kq + tt;  // bits 0..15 half
      float p = 0.f;
      if (bits & (1u << jl)) {
        float e = s1i + s2p[j0 + jl];
        e = (e > 0.f) ? e : 0.2f * e;
        p = __expf(e - Mrow);
      }
      pf[tt] = (__bf16)p;
      jl = 16 + kq + tt;  // bits 16..31 half
      p = 0.f;
      if (bits & (1u << jl)) {
        float e = s1i + s2p[j0 + jl];
        e = (e > 0.f) ? e : 0.2f * e;
        p = __expf(e - Mrow);
      }
      pf[8 + tt] = (__bf16)p;
    }

#pragma unroll
    for (int nt = 0; nt < 8; ++nt) {
      const __bf16* bp = &hT[(nt * 16 + r) * LDS_PAD + half * 8];
      v16bf vf = frag_from_lds(bp, bp + 16);
      acc[nt] = __builtin_amdgcn_wmma_f32_16x16x32_bf16(
          false, pf, false, vf, (short)0, acc[nt], false, false);
    }
    __syncthreads();
  }

  // ---- Epilogue: normalize by 1/L (fetched from the lane holding that row),
  // ELU, store (with head-concat column offset).
  OutT* ob = outp + (long)b * outBatchStride + head * 128;
#pragma unroll
  for (int e = 0; e < 8; ++e) {
    const int row = i0 + half * 8 + e;
    const float il = __shfl(invL, half * 8 + e, 32);  // stats live in lane (row&15)
#pragma unroll
    for (int nt = 0; nt < 8; ++nt) {
      float v = acc[nt][e] * il;
      v = (v > 0.f) ? v : (__expf(v) - 1.f);  // ELU
      ob[(long)row * outRowStride + nt * 16 + r] = (OutT)v;
    }
  }
}

// ---------------------------------------------------------------------------
// K4: gather + 4-head MHA pooling + proj + LayerNorm + ReLU + final proj.
// One 128-thread block per batch element (tiny: ~1.3 MFLOP per block).
// ---------------------------------------------------------------------------
__global__ __launch_bounds__(128)
void mha_head(const float* __restrict__ hfin, const int* __restrict__ cidx,
              const float* __restrict__ wq, const float* __restrict__ bq,
              const float* __restrict__ wk, const float* __restrict__ bk,
              const float* __restrict__ wv, const float* __restrict__ bv,
              const float* __restrict__ wo, const float* __restrict__ bo,
              const float* __restrict__ p1w, const float* __restrict__ p1b,
              const float* __restrict__ lng, const float* __restrict__ lnb,
              const float* __restrict__ p2w, const float* __restrict__ p2b,
              float* __restrict__ outp) {
  __shared__ float q[128], Q[128], ctx[32][128], K[32][128], V[32][128];
  __shared__ float attn[4][32], o1[128], o2[128], p[128], z[128];

  const int b = blockIdx.x, t = threadIdx.x;
  const float* hb = hfin + (long)b * 1024 * 128;

  q[t] = hb[t];  // row 0 = query node
  for (int c = 0; c < 32; ++c) ctx[c][t] = hb[(long)cidx[b * 32 + c] * 128 + t];
  __syncthreads();

  {  // Q projection
    float a = bq[t];
    for (int k = 0; k < 128; ++k) a = fmaf(q[k], wq[k * 128 + t], a);
    Q[t] = a;
  }
  for (int c = 0; c < 32; ++c) {  // K, V projections (thread = output column)
    float ak = bk[t], av = bv[t];
    for (int k = 0; k < 128; ++k) {
      float x = ctx[c][k];
      ak = fmaf(x, wk[k * 128 + t], ak);
      av = fmaf(x, wv[k * 128 + t], av);
    }
    K[c][t] = ak;
    V[c][t] = av;
  }
  __syncthreads();

  {  // scores: thread -> (head = t>>5, key c = t&31)
    const int hh = t >> 5, c = t & 31;
    float s = 0.f;
    for (int d = 0; d < 32; ++d) s = fmaf(Q[hh * 32 + d], K[c][hh * 32 + d], s);
    attn[hh][c] = s * 0.17677669529663687f;  // 1/sqrt(32)
  }
  __syncthreads();
  {  // softmax over the 32 keys of each head
    const int hh = t >> 5, c = t & 31;
    float mx = -3.4e38f;
    for (int d = 0; d < 32; ++d) mx = fmaxf(mx, attn[hh][d]);
    float sm = 0.f;
    for (int d = 0; d < 32; ++d) sm += __expf(attn[hh][d] - mx);
    float v = __expf(attn[hh][c] - mx) / sm;
    __syncthreads();
    attn[hh][c] = v;
  }
  __syncthreads();

  {  // attn @ V, concat heads (column t already belongs to head t>>5)
    const int hh = t >> 5;
    float s = 0.f;
    for (int c = 0; c < 32; ++c) s = fmaf(attn[hh][c], V[c][t], s);
    o1[t] = s;
  }
  __syncthreads();
  {  // out proj
    float a = bo[t];
    for (int k = 0; k < 128; ++k) a = fmaf(o1[k], wo[k * 128 + t], a);
    o2[t] = a;
  }
  __syncthreads();
  {  // p1 proj
    float a = p1b[t];
    for (int k = 0; k < 128; ++k) a = fmaf(o2[k], p1w[k * 128 + t], a);
    p[t] = a;
  }
  __syncthreads();
  {  // LayerNorm + ReLU (redundant per-thread reduction over 128 is cheap)
    float mu = 0.f;
    for (int k = 0; k < 128; ++k) mu += p[k];
    mu *= (1.f / 128.f);
    float var = 0.f;
    for (int k = 0; k < 128; ++k) { float d = p[k] - mu; var = fmaf(d, d, var); }
    var *= (1.f / 128.f);
    float zz = (p[t] - mu) * rsqrtf(var + 1e-5f) * lng[t] + lnb[t];
    z[t] = (zz > 0.f) ? zz : 0.f;
  }
  __syncthreads();
  for (int e = t; e < 768; e += 128) {  // final projection (128 -> 768)
    float a = p2b[e];
    for (int k = 0; k < 128; ++k) a = fmaf(z[k], p2w[k * 768 + e], a);
    outp[(long)b * 768 + e] = a;
  }
}

// ---------------------------------------------------------------------------
// Launch
// ---------------------------------------------------------------------------
extern "C" void kernel_launch(void* const* d_in, const int* in_sizes, int n_in,
                              void* d_out, int out_size, void* d_ws, size_t ws_size,
                              hipStream_t stream) {
  (void)in_sizes; (void)n_in; (void)out_size; (void)ws_size;

  const float* x    = (const float*)d_in[0];
  const int*   adj  = (const int*)d_in[1];
  const int*   cidx = (const int*)d_in[2];
  const float* W1   = (const float*)d_in[3];
  const float* a1_1 = (const float*)d_in[4];
  const float* a2_1 = (const float*)d_in[5];
  const float* W2   = (const float*)d_in[6];
  const float* a1_2 = (const float*)d_in[7];
  const float* a2_2 = (const float*)d_in[8];
  const float* wq   = (const float*)d_in[9];
  const float* bq   = (const float*)d_in[10];
  const float* wk   = (const float*)d_in[11];
  const float* bk   = (const float*)d_in[12];
  const float* wv   = (const float*)d_in[13];
  const float* bv   = (const float*)d_in[14];
  const float* wo   = (const float*)d_in[15];
  const float* bo   = (const float*)d_in[16];
  const float* p1w  = (const float*)d_in[17];
  const float* p1b  = (const float*)d_in[18];
  const float* lng  = (const float*)d_in[19];
  const float* lnb  = (const float*)d_in[20];
  const float* p2w  = (const float*)d_in[21];
  const float* p2b  = (const float*)d_in[22];
  float* outp = (float*)d_out;

  // Workspace layout (bytes, 256-aligned):
  char* ws = (char*)d_ws;
  __bf16*   h1   = (__bf16*)  (ws + 0);         // 8*8*1024*128 bf16 = 16 MB
  float*    s1a  = (float*)   (ws + 16777216);  // 64*1024 f32
  float*    s2a  = (float*)   (ws + 17039360);
  __bf16*   hcat = (__bf16*)  (ws + 17301504);  // 8*1024*1024 bf16 = 16 MB
  __bf16*   h2   = (__bf16*)  (ws + 34078720);  // 8*1024*128 bf16 = 2 MB
  float*    s1b  = (float*)   (ws + 36175872);
  float*    s2b  = (float*)   (ws + 36208640);
  float*    hfin = (float*)   (ws + 36241408);  // 8*1024*128 f32 = 4 MB
  unsigned* adjb = (unsigned*)(ws + 40435712);  // 8*1024*32 u32 = 1 MB

  // adjacency -> bitmask (saves ~0.5 GB of re-read traffic across heads/passes)
  pack_adj<<<dim3(1024), 256, 0, stream>>>(adj, adjb, 8 * 1024 * 32);

  // ---- GAT layer 1 ----
  gemm_wmma_bf16<float><<<dim3(8, 64), 256, 0, stream>>>(
      x, W1, h1, 768, /*aDiv=*/8, /*bMod=*/8,
      (long)1024 * 768, (long)768 * 128, (long)1024 * 128);
  gat_scores<<<dim3(256), 256, 0, stream>>>(h1, a1_1, a2_1, s1a, s2a, 8, 65536);
  gat_attn<__bf16><<<dim3(8, 64), 256, 0, stream>>>(
      h1, s1a, s2a, adjb, hcat, /*H=*/8, (long)1024 * 1024, 1024);

  // ---- GAT layer 2 ----
  gemm_wmma_bf16<__bf16><<<dim3(8, 8), 256, 0, stream>>>(
      hcat, W2, h2, 1024, /*aDiv=*/1, /*bMod=*/1,
      (long)1024 * 1024, 0L, (long)1024 * 128);
  gat_scores<<<dim3(32), 256, 0, stream>>>(h2, a1_2, a2_2, s1b, s2b, 1, 8192);
  gat_attn<float><<<dim3(8, 8), 256, 0, stream>>>(
      h2, s1b, s2b, adjb, hfin, /*H=*/1, (long)1024 * 128, 128);

  // ---- MHA pooling + head ----
  mha_head<<<dim3(8), 128, 0, stream>>>(hfin, cidx, wq, bq, wk, bk, wv, bv,
                                        wo, bo, p1w, p1b, lng, lnb, p2w, p2b, outp);
}
